// Autoformer_11441792876586
// MI455X (gfx1250) — matrix operand order repaired and checked by reading, not compiled, not run on hardware
//
#include <hip/hip_runtime.h>
#include <math.h>

// ---------------- problem constants ----------------
#define BB   32
#define LL   1024
#define CIN  64
#define DD   512
#define HH   8
#define EE   64
#define DFF  2048
#define NLAY 3
#define TOPK 6

typedef __attribute__((ext_vector_type(16))) __bf16 v16bf;
typedef __attribute__((ext_vector_type(8)))  float  v8f;

__device__ __forceinline__ unsigned bfbits(float f) {
  union { float f; unsigned u; } cv; cv.f = f;
  unsigned u = cv.u;
  return (u + 0x7FFFu + ((u >> 16) & 1u)) >> 16;   // RNE fp32->bf16 bits
}
__device__ __forceinline__ unsigned pk2(float lo, float hi) {
  return bfbits(lo) | (bfbits(hi) << 16);
}

// ---------------- WMMA bf16 GEMM: C[M,N] = A[M,K] * W[K,N] (+bias)(+gelu) ----
// BM=128 (8 wave strips of 16 rows), BN=64 (4 col tiles), BK=32, double-buffered LDS
#define BM 128
#define BN 64
#define BK 32

__global__ __launch_bounds__(256)
void gemm_bf16_wmma(const float* __restrict__ A, const float* __restrict__ W,
                    float* __restrict__ C, const float* __restrict__ bias,
                    int M, int N, int K, int act_gelu)
{
  // fragment-order LDS: [buf][strip][laneSlot] -> 16 contiguous bf16 per lane
  __shared__ v16bf AfragL[2][8][32];   // 16 KB
  __shared__ v16bf BfragL[2][4][32];   //  8 KB

  const int tid  = threadIdx.x;
  const int lane = tid & 31;
  const int wave = tid >> 5;      // 0..7 == row strip
  const int half = lane >> 4;
  const int l16  = lane & 15;

  const int m0 = blockIdx.y * BM;
  const int n0 = blockIdx.x * BN;

  v8f acc[4] = {v8f{}, v8f{}, v8f{}, v8f{}};

  const int nk = K / BK;

  float4 aR[4];
  float  bR[8];
  // B staging coords (K-major, coalesced over N)
  const int nl = tid & 63;        // column in tile
  const int rg = tid >> 6;        // 0..3 -> 4 consecutive K rows per half

  // ---- stage tile kt into registers ----
  auto load_regs = [&](int kt) {
    const int k0 = kt * BK;
#pragma unroll
    for (int s = 0; s < 4; ++s) {
      int idx = tid + s * 256;
      int r = idx >> 3, c0 = (idx & 7) << 2;
      aR[s] = *(const float4*)&A[(size_t)(m0 + r) * K + (k0 + c0)];
    }
#pragma unroll
    for (int hf = 0; hf < 2; ++hf)
#pragma unroll
      for (int i = 0; i < 4; ++i) {
        int r = hf * 8 + (rg & 1) * 4 + (rg >> 1) * 16 + i;
        bR[hf * 4 + i] = W[(size_t)(k0 + r) * N + (n0 + nl)];
      }
  };

  // ---- registers -> fragment-order LDS (packed 64-bit stores) ----
  auto store_tile = [&](int buf) {
    unsigned* Au = (unsigned*)&AfragL[buf][0][0];
#pragma unroll
    for (int s = 0; s < 4; ++s) {
      int idx = tid + s * 256;
      int r = idx >> 3, c0 = (idx & 7) << 2;
      int strip = r >> 4, m = r & 15;
      int hf = (c0 >> 3) & 1;
      int vg0 = ((c0 >> 1) & 3) | ((c0 >> 4) << 2);
      int elt = ((strip * 32) + hf * 16 + m) * 16 + (vg0 << 1);
      *(uint2*)&Au[elt >> 1] = make_uint2(pk2(aR[s].x, aR[s].y),
                                          pk2(aR[s].z, aR[s].w));
    }
    unsigned* Bu = (unsigned*)&BfragL[buf][0][0];
    int nstrip = nl >> 4, n = nl & 15;
#pragma unroll
    for (int hf = 0; hf < 2; ++hf) {
      int elt = ((nstrip * 32) + hf * 16 + n) * 16 + rg * 4;
      *(uint2*)&Bu[elt >> 1] = make_uint2(pk2(bR[hf * 4 + 0], bR[hf * 4 + 1]),
                                          pk2(bR[hf * 4 + 2], bR[hf * 4 + 3]));
    }
  };

  // prologue
  load_regs(0);
  store_tile(0);
  __syncthreads();

  for (int kt = 0; kt < nk; ++kt) {
    const int cur = kt & 1;
    if (kt + 1 < nk) load_regs(kt + 1);

    // compute: 1 A fragment, 4 B fragments, 4 WMMA (A reused)
    v16bf af = AfragL[cur][wave][lane];
#pragma unroll
    for (int ns = 0; ns < 4; ++ns) {
      v16bf bfr = BfragL[cur][ns][lane];
      acc[ns] = __builtin_amdgcn_wmma_f32_16x16x32_bf16(
          false, af, false, bfr, (short)0, acc[ns], false, false);
    }

    if (kt + 1 < nk) store_tile(cur ^ 1);
    __syncthreads();
  }

  // epilogue: bias + exact GELU fused
#pragma unroll
  for (int ns = 0; ns < 4; ++ns) {
    int ncol = n0 + ns * 16 + l16;
    float bia = bias ? bias[ncol] : 0.f;
#pragma unroll
    for (int r = 0; r < 8; ++r) {
      int mrow = m0 + wave * 16 + r + (half << 3);
      float vv = acc[ns][r] + bia;
      if (act_gelu) vv = 0.5f * vv * (1.0f + erff(vv * 0.70710678118f));
      C[(size_t)mrow * N + ncol] = vv;
    }
  }
}

// ---------------- misc elementwise ----------------
__global__ void zero_f(float* p, int n) {
  int i = blockIdx.x * blockDim.x + threadIdx.x;
  if (i < n) p[i] = 0.f;
}

// emb input: out[b,l,j*64+c] = x[b,(l-1+j) mod L, c]
__global__ void build_emb(const float* __restrict__ x, float* __restrict__ out)
{
  size_t idx = (size_t)blockIdx.x * blockDim.x + threadIdx.x;
  if (idx >= (size_t)BB * LL * 192) return;
  int c = (int)(idx % 64);
  int j = (int)((idx / 64) % 3);
  int l = (int)((idx / 192) % LL);
  int b = (int)(idx / ((size_t)192 * LL));
  int ls = (l - 1 + j + LL) & (LL - 1);
  out[idx] = x[(((size_t)b * LL) + ls) * CIN + c];
}

// tiled transpose (B,L,D) -> (B,D,L) so FFT rows stream coalesced
__global__ __launch_bounds__(256)
void transpose_LD(const float* __restrict__ in, float* __restrict__ out)
{
  __shared__ float t[32][33];
  const int b  = blockIdx.z;
  const int l0 = blockIdx.x * 32, d0 = blockIdx.y * 32;
  const int lx = threadIdx.x & 31, ly = threadIdx.x >> 5;   // 32 x 8
  for (int i = ly; i < 32; i += 8)
    t[i][lx] = in[((size_t)b * LL + l0 + i) * DD + d0 + lx];
  __syncthreads();
  for (int i = ly; i < 32; i += 8)
    out[((size_t)b * DD + d0 + i) * LL + l0 + lx] = t[lx][i];
}

// ---------------- FFT autocorrelation path ----------------
// qT,kT are (B,D,L); one packed complex FFT per (b,d) row; accumulate cross spectrum.
__global__ __launch_bounds__(256)
void fft_cross_accum(const float* __restrict__ qT, const float* __restrict__ kT,
                     float* __restrict__ spec /* B x 513 x 2 */)
{
  __shared__ float2 z[LL];
  __shared__ float2 acc[513];
  const int tid = threadIdx.x;
  const int b = blockIdx.y;
  const int chunk = blockIdx.x;          // 8 chunks of 64 rows

  for (int i = tid; i < 513; i += 256) acc[i] = make_float2(0.f, 0.f);
  __syncthreads();

  for (int rr = chunk * 64; rr < chunk * 64 + 64; ++rr) {
    const float* qp = qT + ((size_t)b * DD + rr) * LL;
    const float* kp = kT + ((size_t)b * DD + rr) * LL;
    for (int l = tid; l < LL; l += 256)
      z[l] = make_float2(qp[l], kp[l]);
    __syncthreads();
    for (int i = tid; i < LL; i += 256) {
      int j = (int)(__brev((unsigned)i) >> 22);
      if (i < j) { float2 t = z[i]; z[i] = z[j]; z[j] = t; }
    }
    __syncthreads();
    for (int len = 2; len <= LL; len <<= 1) {
      int hl = len >> 1;
      float ang = -6.283185307179586f / (float)len;
      for (int t = tid; t < LL / 2; t += 256) {
        int g = t / hl, p = t % hl;
        int i0 = g * len + p, i1 = i0 + hl;
        float s, c; __sincosf(ang * (float)p, &s, &c);
        float2 u = z[i0], vv = z[i1];
        float2 w = make_float2(c * vv.x - s * vv.y, c * vv.y + s * vv.x);
        z[i0] = make_float2(u.x + w.x, u.y + w.y);
        z[i1] = make_float2(u.x - w.x, u.y - w.y);
      }
      __syncthreads();
    }
    for (int f = tid; f <= 512; f += 256) {
      float2 zf = z[f];
      float2 zn = z[(LL - f) & (LL - 1)];
      float qr = 0.5f * (zf.x + zn.x), qi = 0.5f * (zf.y - zn.y);
      float dr = 0.5f * (zf.x - zn.x), di = 0.5f * (zf.y + zn.y);
      float kr = di, ki = -dr;                     // KF = (zf-conj(zn))/(2i)
      acc[f].x += qr * kr + qi * ki;
      acc[f].y += qi * kr - qr * ki;
    }
    __syncthreads();
  }
  for (int f = tid; f <= 512; f += 256) {
    atomicAdd(&spec[((size_t)b * 513 + f) * 2 + 0], acc[f].x);
    atomicAdd(&spec[((size_t)b * 513 + f) * 2 + 1], acc[f].y);
  }
}

__global__ __launch_bounds__(256)
void ifft_corr(const float* __restrict__ spec, float* __restrict__ corr)
{
  __shared__ float2 z[LL];
  const int tid = threadIdx.x, b = blockIdx.x;
  for (int f = tid; f < LL; f += 256) {
    float re, im;
    if (f <= 512) {
      re = spec[((size_t)b * 513 + f) * 2];
      im = spec[((size_t)b * 513 + f) * 2 + 1];
    } else {
      re =  spec[((size_t)b * 513 + (LL - f)) * 2];
      im = -spec[((size_t)b * 513 + (LL - f)) * 2 + 1];
    }
    z[f] = make_float2(re, im);
  }
  __syncthreads();
  for (int i = tid; i < LL; i += 256) {
    int j = (int)(__brev((unsigned)i) >> 22);
    if (i < j) { float2 t = z[i]; z[i] = z[j]; z[j] = t; }
  }
  __syncthreads();
  for (int len = 2; len <= LL; len <<= 1) {
    int hl = len >> 1;
    float ang = 6.283185307179586f / (float)len;
    for (int t = tid; t < LL / 2; t += 256) {
      int g = t / hl, p = t % hl;
      int i0 = g * len + p, i1 = i0 + hl;
      float s, c; __sincosf(ang * (float)p, &s, &c);
      float2 u = z[i0], vv = z[i1];
      float2 w = make_float2(c * vv.x - s * vv.y, c * vv.y + s * vv.x);
      z[i0] = make_float2(u.x + w.x, u.y + w.y);
      z[i1] = make_float2(u.x - w.x, u.y - w.y);
    }
    __syncthreads();
  }
  const float scale = 1.0f / ((float)LL * (float)(HH * EE));
  for (int l = tid; l < LL; l += 256)
    corr[(size_t)b * LL + l] = z[l].x * scale;
}

__global__ __launch_bounds__(256)
void topk_softmax(const float* __restrict__ corr, float* __restrict__ tc,
                  int* __restrict__ delays)
{
  __shared__ float vals[LL];
  __shared__ float rv[256];
  __shared__ int   ri[256];
  __shared__ float wsel[TOPK];
  __shared__ int   isel[TOPK];
  const int tid = threadIdx.x, b = blockIdx.x;
  for (int i = tid; i < LL; i += 256) vals[i] = corr[(size_t)b * LL + i];
  __syncthreads();
  for (int kk = 0; kk < TOPK; ++kk) {
    float best = -1e30f; int bi = 0;
    for (int i = tid; i < LL; i += 256)
      if (vals[i] > best) { best = vals[i]; bi = i; }
    rv[tid] = best; ri[tid] = bi;
    __syncthreads();
    for (int s = 128; s > 0; s >>= 1) {
      if (tid < s && rv[tid + s] > rv[tid]) { rv[tid] = rv[tid + s]; ri[tid] = ri[tid + s]; }
      __syncthreads();
    }
    if (tid == 0) { wsel[kk] = rv[0]; isel[kk] = ri[0]; vals[ri[0]] = -1e30f; }
    __syncthreads();
  }
  if (tid == 0) {
    float mx = wsel[0];
    for (int kk = 1; kk < TOPK; ++kk) mx = fmaxf(mx, wsel[kk]);
    float e[TOPK]; float sum = 0.f;
    for (int kk = 0; kk < TOPK; ++kk) { e[kk] = __expf(wsel[kk] - mx); sum += e[kk]; }
    for (int kk = 0; kk < TOPK; ++kk) {
      tc[b * TOPK + kk] = e[kk] / sum;
      delays[b * TOPK + kk] = isel[kk];
    }
  }
}

__global__ void roll_gather(const float* __restrict__ v, const float* __restrict__ tc,
                            const int* __restrict__ delays, float* __restrict__ out)
{
  size_t idx = (size_t)blockIdx.x * blockDim.x + threadIdx.x;
  if (idx >= (size_t)BB * LL * DD) return;
  int d = (int)(idx & (DD - 1));
  int l = (int)((idx >> 9) & (LL - 1));
  int b = (int)(idx >> 19);
  float s = 0.f;
#pragma unroll
  for (int kk = 0; kk < TOPK; ++kk) {
    int dl = delays[b * TOPK + kk];
    int ls = (l + dl) & (LL - 1);
    s += tc[b * TOPK + kk] * v[(((size_t)b * LL + ls) << 9) + d];
  }
  out[idx] = s;
}

// series decomp: out = (x+add) - movavg_25(x+add), edge-replicated, running window
__global__ void decomp_ma(const float* __restrict__ x, const float* __restrict__ add,
                          float* __restrict__ out)
{
  int t = blockIdx.x * blockDim.x + threadIdx.x;     // over B*D
  if (t >= BB * DD) return;
  int d = t & (DD - 1), b = t >> 9;
  const size_t base = (size_t)b * LL * DD;
  auto S = [&](int l) -> float {
    l = l < 0 ? 0 : (l > LL - 1 ? LL - 1 : l);
    size_t o = base + ((size_t)l << 9) + d;
    float vv = x[o];
    if (add) vv += add[o];
    return vv;
  };
  float wsum = 0.f;
  for (int j = -12; j <= 12; ++j) wsum += S(j);
  for (int l = 0; l < LL; ++l) {
    out[base + ((size_t)l << 9) + d] = S(l) - wsum * (1.0f / 25.0f);
    wsum += S(l + 13) - S(l - 12);
  }
}

__global__ __launch_bounds__(128)
void row_layernorm(const float* __restrict__ x, const float* __restrict__ g,
                   const float* __restrict__ be, float* __restrict__ out)
{
  __shared__ float red[128];
  const int row = blockIdx.x, tid = threadIdx.x;
  const float* xr = x + (size_t)row * DD;
  float s = 0.f;
  for (int i = tid; i < DD; i += 128) s += xr[i];
  red[tid] = s; __syncthreads();
  for (int st = 64; st > 0; st >>= 1) { if (tid < st) red[tid] += red[tid + st]; __syncthreads(); }
  float mu = red[0] / (float)DD;
  __syncthreads();
  float v = 0.f;
  for (int i = tid; i < DD; i += 128) { float dd = xr[i] - mu; v += dd * dd; }
  red[tid] = v; __syncthreads();
  for (int st = 64; st > 0; st >>= 1) { if (tid < st) red[tid] += red[tid + st]; __syncthreads(); }
  float inv = rsqrtf(red[0] / (float)DD + 1e-5f);
  for (int i = tid; i < DD; i += 128)
    out[(size_t)row * DD + i] = (xr[i] - mu) * inv * g[i] + be[i];
}

__global__ void colmean_sub(float* __restrict__ x)
{
  int t = blockIdx.x * blockDim.x + threadIdx.x;
  if (t >= BB * DD) return;
  int d = t & (DD - 1), b = t >> 9;
  size_t base = (size_t)b * LL * DD + d;
  float s = 0.f;
  for (int l = 0; l < LL; ++l) s += x[base + (size_t)l * DD];
  float mu = s / (float)LL;
  for (int l = 0; l < LL; ++l) x[base + (size_t)l * DD] -= mu;
}

// ---------------- host orchestration ----------------
static inline void launch_gemm(const float* A, const float* W, float* C,
                               const float* bias, int M, int N, int K,
                               int gelu, hipStream_t s)
{
  dim3 g(N / BN, M / BM);
  gemm_bf16_wmma<<<g, 256, 0, s>>>(A, W, C, bias, M, N, K, gelu);
}

extern "C" void kernel_launch(void* const* d_in, const int* in_sizes, int n_in,
                              void* d_out, int out_size, void* d_ws, size_t ws_size,
                              hipStream_t stream)
{
  (void)in_sizes; (void)n_in; (void)out_size; (void)ws_size;
  const float* x       = (const float*)d_in[0];
  const float* embed_w = (const float*)d_in[1];   // (3,64,512) == (192,512) row-major
  const float* wq      = (const float*)d_in[2];
  const float* bq      = (const float*)d_in[3];
  const float* wk      = (const float*)d_in[4];
  const float* bk      = (const float*)d_in[5];
  const float* wv      = (const float*)d_in[6];
  const float* bv      = (const float*)d_in[7];
  const float* wo      = (const float*)d_in[8];
  const float* bo      = (const float*)d_in[9];
  const float* w1      = (const float*)d_in[10];
  const float* w2      = (const float*)d_in[11];
  const float* ln_g    = (const float*)d_in[12];
  const float* ln_b    = (const float*)d_in[13];
  const float* proj_w  = (const float*)d_in[14];
  const float* proj_b  = (const float*)d_in[15];
  float* out = (float*)d_out;

  float* ws = (float*)d_ws;
  const size_t SZ_H  = (size_t)BB * LL * DD;     // 16.78M floats
  const size_t SZ_FF = (size_t)BB * LL * DFF;    // 67.11M floats
  float* hA   = ws;                  // h ping
  float* bufV = ws + 1 * SZ_H;
  float* bufQ = ws + 2 * SZ_H;       // q, later a@wo output
  float* bufK = ws + 3 * SZ_H;
  float* bufA = ws + 4 * SZ_H;       // attn out / ffn out
  float* hB   = ws + 5 * SZ_H;       // h pong
  float* bufF = ws + 6 * SZ_H;       // DFF intermediate; emb input; qT/kT scratch
  float* spec = bufF + SZ_FF;                         // B*513*2
  float* corr = spec + (size_t)BB * 513 * 2;          // B*L
  float* tc   = corr + (size_t)BB * LL;               // B*6
  int*   dly  = (int*)(tc + BB * TOPK);               // B*6

  const int M = BB * LL;
  const dim3 tgrid(LL / 32, DD / 32, BB);

  // 1) token embedding: circular k=3 conv as GEMM (M x 192)(192 x 512)
  {
    size_t tot = (size_t)BB * LL * 192;
    build_emb<<<(int)((tot + 255) / 256), 256, 0, stream>>>(x, bufF);
    launch_gemm(bufF, embed_w, hA, nullptr, M, DD, 192, 0, stream);
  }

  float* hcur = hA;
  float* hnext = hB;

  for (int l = 0; l < NLAY; ++l) {
    const float* wq_l = wq + (size_t)l * DD * DD;
    const float* wk_l = wk + (size_t)l * DD * DD;
    const float* wv_l = wv + (size_t)l * DD * DD;
    const float* wo_l = wo + (size_t)l * DD * DD;
    const float* w1_l = w1 + (size_t)l * DD * DFF;
    const float* w2_l = w2 + (size_t)l * DFF * DD;

    // QKV projections
    launch_gemm(hcur, wq_l, bufQ, bq + l * DD, M, DD, DD, 0, stream);
    launch_gemm(hcur, wk_l, bufK, bk + l * DD, M, DD, DD, 0, stream);
    launch_gemm(hcur, wv_l, bufV, bv + l * DD, M, DD, DD, 0, stream);

    // AutoCorrelation: transpose q,k for coalesced FFT rows, sum cross-spectra
    float* qT = bufF;
    float* kT = bufF + SZ_H;
    transpose_LD<<<tgrid, 256, 0, stream>>>(bufQ, qT);
    transpose_LD<<<tgrid, 256, 0, stream>>>(bufK, kT);
    zero_f<<<(BB * 513 * 2 + 255) / 256, 256, 0, stream>>>(spec, BB * 513 * 2);
    fft_cross_accum<<<dim3(8, BB), 256, 0, stream>>>(qT, kT, spec);
    ifft_corr<<<BB, 256, 0, stream>>>(spec, corr);
    topk_softmax<<<BB, 256, 0, stream>>>(corr, tc, dly);
    {
      size_t tot = (size_t)M * DD;
      roll_gather<<<(int)((tot + 255) / 256), 256, 0, stream>>>(bufV, tc, dly, bufA);
    }
    // output projection (a @ wo + bo) -> bufQ
    launch_gemm(bufA, wo_l, bufQ, bo + l * DD, M, DD, DD, 0, stream);

    // decomp1: hnext = (hcur + bufQ) - movavg
    decomp_ma<<<(BB * DD + 255) / 256, 256, 0, stream>>>(hcur, bufQ, hnext);
    { float* t = hcur; hcur = hnext; hnext = t; }

    // FFN: gelu(h @ w1) @ w2 (no biases), GELU fused into GEMM1 epilogue
    launch_gemm(hcur, w1_l, bufF, nullptr, M, DFF, DD, 1, stream);
    launch_gemm(bufF, w2_l, bufA, nullptr, M, DD, DFF, 0, stream);

    // decomp2
    decomp_ma<<<(BB * DD + 255) / 256, 256, 0, stream>>>(hcur, bufA, hnext);
    { float* t = hcur; hcur = hnext; hnext = t; }
  }

  // my_Layernorm -> bufQ, then subtract time-mean in place
  row_layernorm<<<M, 128, 0, stream>>>(hcur, ln_g, ln_b, bufQ);
  colmean_sub<<<(BB * DD + 255) / 256, 256, 0, stream>>>(bufQ);

  // final projection (M x 512)(512 x 64) + bias -> out
  launch_gemm(bufQ, proj_w, out, proj_b, M, CIN, DD, 0, stream);
}
